// Node2Edge_5557687681587
// MI455X (gfx1250) — matrix operand order repaired and checked by reading, not compiled, not run on hardware
//
#include <hip/hip_runtime.h>
#include <hip/hip_bf16.h>

// Edge-MLP gather-GEMM for MI455X (gfx1250, wave32, WMMA).
//   out[e] = x[src_i[e]] @ W[0:128] + x[src_j[e]] @ W[128:256] + ea[e] @ W[256:272] + b
// M=600000, K=272 (padded to 288), N=128, fp32 reference.
// bf16 hi/lo split (3 WMMAs per 16x16x32 step) ~ fp32 accuracy on the bf16 WMMA
// fast path; W^T staged hi/lo in LDS (144 KB); A gathered from global (x is
// L2-resident: 51 MB < 192 MB L2). K-chunk loops kept at unroll(1) to bound the
// live VGPR set (~120) and avoid the scratch spills seen with full unrolling.

#define NEDGES    600000
#define NODE_DIM  128
#define EDGE_DIM  16
#define ODIM      128
#define K_TOTAL   (2 * NODE_DIM + EDGE_DIM)   // 272
#define K_PAD     288                         // 9 chunks of 32

#define BLOCK     256
#define WAVES     (BLOCK / 32)                // 8 wave32
#define EDGES_PER_BLOCK (WAVES * 16)          // 128 edges / block-tile

#define LDS_HALF  (ODIM * K_PAD)              // bf16 elements per copy (36864)
#define LDS_BYTES (2 * LDS_HALF * 2)          // hi + lo = 147456 B (< 320 KB/WGP)

typedef __attribute__((ext_vector_type(16))) __bf16 v16bf;
typedef __attribute__((ext_vector_type(8)))  float  v8f;

// One K-chunk of 32: convert 16 gathered fp32 into bf16 hi/lo A-fragments and
// accumulate 8 N-tiles with the 3-product split.
__device__ __forceinline__ void kchunk_step(const v8f& g0, const v8f& g1, int c,
                                            const __bf16* __restrict__ wt_hi,
                                            const __bf16* __restrict__ wt_lo,
                                            int col, int half, v8f acc[8])
{
    v16bf ahi, alo;
    #pragma unroll
    for (int q = 0; q < 8; ++q) {
        float  f0 = g0[q], f1 = g1[q];
        __bf16 h0 = (__bf16)f0, h1 = (__bf16)f1;
        ahi[q]     = h0;  alo[q]     = (__bf16)(f0 - (float)h0);
        ahi[q + 8] = h1;  alo[q + 8] = (__bf16)(f1 - (float)h1);
    }
    #pragma unroll
    for (int nt = 0; nt < 8; ++nt) {
        // B layout (32x16 bf16): lane%16 = N, lane half h holds
        // K h*8..h*8+7 and 16+h*8..16+h*8+7 of the W^T row (matches A).
        const int boff = (nt * 16 + col) * K_PAD + c * 32 + half * 8;
        typedef __attribute__((ext_vector_type(8))) __bf16 v8bf;
        v8bf bh0 = *(const v8bf*)(wt_hi + boff), bh1 = *(const v8bf*)(wt_hi + boff + 16);
        v8bf bl0 = *(const v8bf*)(wt_lo + boff), bl1 = *(const v8bf*)(wt_lo + boff + 16);
        v16bf bhi = __builtin_shufflevector(bh0, bh1, 0, 1, 2, 3, 4, 5, 6, 7, 8, 9, 10, 11, 12, 13, 14, 15);
        v16bf blo = __builtin_shufflevector(bl0, bl1, 0, 1, 2, 3, 4, 5, 6, 7, 8, 9, 10, 11, 12, 13, 14, 15);
        acc[nt] = __builtin_amdgcn_wmma_f32_16x16x32_bf16(
            false, ahi, false, bhi, (short)0, acc[nt], false, false);
        acc[nt] = __builtin_amdgcn_wmma_f32_16x16x32_bf16(
            false, alo, false, bhi, (short)0, acc[nt], false, false);
        acc[nt] = __builtin_amdgcn_wmma_f32_16x16x32_bf16(
            false, ahi, false, blo, (short)0, acc[nt], false, false);
    }
}

__global__ __launch_bounds__(BLOCK, 1)
void edge_mlp_wmma(const float* __restrict__ x,
                   const int*   __restrict__ eidx,   // [2][NEDGES]
                   const float* __restrict__ eattr,  // [NEDGES][16]
                   const float* __restrict__ W,      // [272][128]
                   const float* __restrict__ bias,   // [128]
                   float*       __restrict__ out,    // [NEDGES][128]
                   int numTiles)
{
    extern __shared__ __bf16 lds[];
    __bf16* wt_hi = lds;             // W^T hi: [n][k], n-major, K contiguous
    __bf16* wt_lo = lds + LDS_HALF;  // W^T lo

    const int tid = threadIdx.x;

    // ---- Stage W^T into LDS as bf16 hi/lo, zero-pad K to 288 ----
    #pragma unroll 1
    for (int i = tid; i < K_TOTAL * ODIM; i += BLOCK) {     // coalesced read of W
        int k = i >> 7;          // i / 128
        int n = i & 127;
        float  f = W[i];
        __bf16 h = (__bf16)f;
        wt_hi[n * K_PAD + k] = h;
        wt_lo[n * K_PAD + k] = (__bf16)(f - (float)h);
    }
    #pragma unroll 1
    for (int i = tid; i < (K_PAD - K_TOTAL) * ODIM; i += BLOCK) {
        int k = K_TOTAL + (i >> 7);
        int n = i & 127;
        wt_hi[n * K_PAD + k] = (__bf16)0.0f;
        wt_lo[n * K_PAD + k] = (__bf16)0.0f;
    }
    __syncthreads();

    const int lane = tid & 31;
    const int wave = tid >> 5;       // wave32 on gfx1250
    const int half = lane >> 4;      // lane group (0: lanes 0-15, 1: lanes 16-31)
    const int col  = lane & 15;      // N within a 16x16 tile / A-row within wave

    #pragma unroll 1
    for (int tile = blockIdx.x; tile < numTiles; tile += gridDim.x) {
        const long waveBase = (long)tile * EDGES_PER_BLOCK + wave * 16;

        // Prefetch next tile's edge indices (gfx1250 global_prefetch path).
        {
            long nb = ((long)tile + gridDim.x) * EDGES_PER_BLOCK;
            if (nb < NEDGES) __builtin_prefetch(&eidx[nb + lane], 0, 1);
        }

        // This lane supplies A-row r = lane%16 of the wave's 16-edge tile.
        long eRow = waveBase + col;
        long eCl  = eRow < NEDGES ? eRow : (NEDGES - 1);   // clamp (stores predicated)
        const int si = eidx[eCl];
        const int sj = eidx[NEDGES + eCl];
        const float* rowI = x + (long)si * NODE_DIM;
        const float* rowJ = x + (long)sj * NODE_DIM;
        const float* rowE = eattr + eCl * EDGE_DIM;

        // Accumulators: 8 N-tiles of 16x16 f32, init with bias (C layout: lane%16 = N).
        v8f acc[8];
        #pragma unroll
        for (int nt = 0; nt < 8; ++nt) {
            float bv = bias[nt * 16 + col];
            v8f a;
            #pragma unroll
            for (int q = 0; q < 8; ++q) a[q] = bv;
            acc[nt] = a;
        }

        // ---- K sweep. A layout (16-bit A 16x32): lane holds
        //      K = [h*8 .. h*8+7] and [16+h*8 .. 16+h*8+7] of its row. ----
        #pragma unroll 1
        for (int c = 0; c < 4; ++c) {            // chunks 0-3: x[src_i]
            v8f g0 = *(const v8f*)(rowI + c * 32 + half * 8);
            v8f g1 = *(const v8f*)(rowI + c * 32 + 16 + half * 8);
            kchunk_step(g0, g1, c, wt_hi, wt_lo, col, half, acc);
        }
        #pragma unroll 1
        for (int c = 4; c < 8; ++c) {            // chunks 4-7: x[src_j]
            v8f g0 = *(const v8f*)(rowJ + (c - 4) * 32 + half * 8);
            v8f g1 = *(const v8f*)(rowJ + (c - 4) * 32 + 16 + half * 8);
            kchunk_step(g0, g1, c, wt_hi, wt_lo, col, half, acc);
        }
        {                                        // chunk 8: edge_attr + zero pad
            v8f g0 = *(const v8f*)(rowE + half * 8);   // K 256..271 (16 real cols)
            v8f g1;
            #pragma unroll
            for (int q = 0; q < 8; ++q) g1[q] = 0.0f;  // K 272..287 pad
            kchunk_step(g0, g1, 8, wt_hi, wt_lo, col, half, acc);
        }

        // ---- Store: C/D layout: VGPR v -> M = v + half*8, lane%16 = N ----
        #pragma unroll
        for (int v = 0; v < 8; ++v) {
            long eo = waveBase + half * 8 + v;
            if (eo < NEDGES) {
                float* op = out + eo * ODIM + col;
                #pragma unroll
                for (int nt = 0; nt < 8; ++nt) op[nt * 16] = acc[nt][v];
            }
        }
    }
}

extern "C" void kernel_launch(void* const* d_in, const int* in_sizes, int n_in,
                              void* d_out, int out_size, void* d_ws, size_t ws_size,
                              hipStream_t stream) {
    const float* x     = (const float*)d_in[0];
    const int*   eidx  = (const int*)d_in[1];   // harness: integer -> const int*
    const float* eattr = (const float*)d_in[2];
    const float* W     = (const float*)d_in[3];
    const float* bias  = (const float*)d_in[4];
    float*       out   = (float*)d_out;

    const int numTiles = (NEDGES + EDGES_PER_BLOCK - 1) / EDGES_PER_BLOCK;  // 4688
    const int grid = numTiles < 2048 ? numTiles : 2048;  // persistent: amortize W staging

    hipFuncSetAttribute((const void*)edge_mlp_wmma,
                        hipFuncAttributeMaxDynamicSharedMemorySize, LDS_BYTES);
    edge_mlp_wmma<<<grid, BLOCK, LDS_BYTES, stream>>>(x, eidx, eattr, W, bias, out,
                                                      numTiles);
}
